// ACARHead_22539988369546
// MI455X (gfx1250) — matrix operand order repaired and checked by reading, not compiled, not run on hardware
//
#include <hip/hip_runtime.h>
#include <hip/hip_bf16.h>
#include <cstdint>

// ---------------- CDNA5 WMMA types ----------------
typedef __attribute__((ext_vector_type(16))) __bf16       v16bf;
typedef __attribute__((ext_vector_type(8)))  float        v8f;
typedef __attribute__((ext_vector_type(4)))  unsigned int v4u;

union FragU { v4u u[2]; v16bf v; };

__device__ __forceinline__ __bf16 f2bf(float f) {
  unsigned u = __builtin_bit_cast(unsigned, f);
  unsigned r = u + 0x7FFFu + ((u >> 16) & 1u);   // round-to-nearest-even
  unsigned short h = (unsigned short)(r >> 16);
  return __builtin_bit_cast(__bf16, h);
}

__device__ __forceinline__ v8f wmma_bf16(v16bf a, v16bf b, v8f c) {
  // D = A(16x32 bf16) * B(32x16 bf16) + C(16x16 f32)
  return __builtin_amdgcn_wmma_f32_16x16x32_bf16(false, a, false, b, (short)0, c, false, false);
}

// ---------------- problem constants ----------------
#define BB     8
#define RR     10
#define NROI   80      // B*R
#define WIDTH_ 2048
#define TT     4
#define HH     16
#define WW     22
#define HW_    352     // 16*22
#define RED    1024
#define HID_   512
#define NCLS_  80
#define CH_    14      // conv2 valid out H
#define CW_    20      // conv2 valid out W
#define CHW_   280
#define PH_    7       // pooled H
#define PW_    10      // pooled W
#define PHW_   70

// =====================================================================
// Pre-swizzle a (K x N) matrix (arbitrary element strides sk, sn) into
// WMMA B-fragment layout: frag[kt][nt][lane][16]
//   k = kt*32 + (lane>>4)*16 + i ,  n = nt*16 + (lane&15)
// =====================================================================
__global__ void k_prep_frag(const float* __restrict__ src, __bf16* __restrict__ dst,
                            int K, int N, int ntiles, int sk, int sn) {
  int idx = blockIdx.x * blockDim.x + threadIdx.x;
  int total = (K >> 5) * ntiles * 512;
  if (idx >= total) return;
  int i    = idx & 15;
  int lane = (idx >> 4) & 31;
  int t    = idx >> 9;
  int nt   = t % ntiles;
  int kt   = t / ntiles;
  int k = kt * 32 + ((lane >> 4) << 4) + i;
  int n = nt * 16 + (lane & 15);
  float v = (k < K && n < N) ? src[(size_t)k * sk + (size_t)n * sn] : 0.f;
  dst[idx] = f2bf(v);
}

// =====================================================================
// temporal mean of x (B,2048,4,16,22) -> bf16 A matrix [b*352+hw][c]
// LDS transpose tile so reads (f32, hw-contig) and writes (bf16, c-contig)
// are both 128B coalesced.  grid (352/32, 2048/64, 8), block 256.
// =====================================================================
__global__ __launch_bounds__(256) void k_meanT(const float* __restrict__ x,
                                               __bf16* __restrict__ xm) {
  __shared__ float lds[64][33];
  const int b   = blockIdx.z;
  const int c0  = blockIdx.y * 64;
  const int hw0 = blockIdx.x * 32;
  for (int i = threadIdx.x; i < 64 * 32; i += 256) {
    int cl = i >> 5, hl = i & 31;
    const float* p = x + ((size_t)(b * WIDTH_ + c0 + cl)) * (TT * HW_) + hw0 + hl;
    lds[cl][hl] = (p[0] + p[HW_] + p[2 * HW_] + p[3 * HW_]) * 0.25f;
  }
  __syncthreads();
  for (int i = threadIdx.x; i < 64 * 32; i += 256) {
    int hl = i >> 6, cl = i & 63;
    xm[((size_t)(b * HW_ + hw0 + hl)) * WIDTH_ + c0 + cl] = f2bf(lds[cl][hl]);
  }
}

// =====================================================================
// bf16 GEMM  C[M,N] = A[M,K] * B[K,N]  (B pre-swizzled fragments)
// block = 128 threads (4 waves); each wave owns a 32x64 C tile:
//   2 A-fragments x 4 B-fragments -> 8 wmma per K-step, all fragment
//   loads issued before the wmma burst (single wait per K-step).
// =====================================================================
__global__ __launch_bounds__(128) void k_gemm(
    const __bf16* __restrict__ A, const __bf16* __restrict__ Bf,
    float* __restrict__ Cf, __bf16* __restrict__ Cb,
    int M, int N, int K, int lda, int ldc, int ntiles, int relu) {
  const int lane  = threadIdx.x & 31;
  const int wave  = threadIdx.x >> 5;
  const int mBase = blockIdx.x * 128 + wave * 32;
  const int nt0   = blockIdx.y * 4;
  int row0 = mBase + (lane & 15);       if (row0 >= M) row0 = M - 1;
  int row1 = mBase + 16 + (lane & 15);  if (row1 >= M) row1 = M - 1;
  const int kOff = (lane >> 4) * 16;
  const __bf16* arow0 = A + (size_t)row0 * lda + kOff;
  const __bf16* arow1 = A + (size_t)row1 * lda + kOff;

  v8f z = {0.f,0.f,0.f,0.f,0.f,0.f,0.f,0.f};
  v8f acc[8];
#pragma unroll
  for (int i = 0; i < 8; ++i) acc[i] = z;

  const int kts = K >> 5;
  for (int kt = 0; kt < kts; ++kt) {
    FragU a0, a1, b[4];
    const v4u* pa0 = (const v4u*)(arow0 + kt * 32);
    const v4u* pa1 = (const v4u*)(arow1 + kt * 32);
    a0.u[0] = pa0[0]; a0.u[1] = pa0[1];       // 16 contiguous K per lane half
    a1.u[0] = pa1[0]; a1.u[1] = pa1[1];
    const __bf16* bbase = Bf + ((size_t)kt * ntiles) * 512 + lane * 16;
#pragma unroll
    for (int t = 0; t < 4; ++t) {
      const v4u* pb = (const v4u*)(bbase + (size_t)(nt0 + t) * 512);
      b[t].u[0] = pb[0]; b[t].u[1] = pb[1];
    }
    // speculative A-stream prefetch (global_prefetch_b8); OOB is dropped
    __builtin_prefetch(arow0 + (kt + 8) * 32, 0, 1);
    __builtin_prefetch(arow1 + (kt + 8) * 32, 0, 1);
#pragma unroll
    for (int t = 0; t < 4; ++t) {
      acc[t]     = wmma_bf16(a0.v, b[t].v, acc[t]);
      acc[4 + t] = wmma_bf16(a1.v, b[t].v, acc[4 + t]);
    }
  }

  const int col  = lane & 15;
  const int rOff = (lane >> 4) * 8;
#pragma unroll
  for (int h = 0; h < 2; ++h) {
#pragma unroll
    for (int t = 0; t < 4; ++t) {
      const int gn = (nt0 + t) * 16 + col;
#pragma unroll
      for (int r = 0; r < 8; ++r) {
        const int gm = mBase + h * 16 + rOff + r;
        if (gm < M && gn < N) {
          float v = acc[h * 4 + t][r];
          if (relu) v = fmaxf(v, 0.f);
          if (Cf) Cf[(size_t)gm * ldc + gn] = v;
          if (Cb) Cb[(size_t)gm * ldc + gn] = f2bf(v);
        }
      }
    }
  }
}

// =====================================================================
// 3x3 conv (implicit GEMM, NHWC bf16 in, fragments for weights [tap][C][O])
// one wave per block; wave = 32 output pixels x 64 output channels
// grid: (ceil(M/32), O/64, nSamples).  pad = 0 (VALID) or 1 (SAME, zero-pad)
// optional relu / f32-residual add; writes f32 and optional bf16 NHWC out
// =====================================================================
__global__ __launch_bounds__(32) void k_conv3x3(
    const __bf16* __restrict__ in, const __bf16* __restrict__ Wf,
    float* __restrict__ outF, __bf16* __restrict__ outB,
    const float* __restrict__ resid,
    int Hin, int Win, int C, int Hout, int Wout, int O,
    int ntiles, int pad, int relu) {
  const int lane = threadIdx.x;
  const int n    = blockIdx.z;
  const int M    = Hout * Wout;
  const int mBase = blockIdx.x * 32;
  const int nt0   = blockIdx.y * 4;
  int row0 = mBase + (lane & 15);       if (row0 >= M) row0 = M - 1;
  int row1 = mBase + 16 + (lane & 15);  if (row1 >= M) row1 = M - 1;
  const int oy0 = row0 / Wout, ox0 = row0 % Wout;
  const int oy1 = row1 / Wout, ox1 = row1 % Wout;
  const int kOff = (lane >> 4) * 16;
  const __bf16* inN = in + (size_t)n * Hin * Win * C;

  v8f z = {0.f,0.f,0.f,0.f,0.f,0.f,0.f,0.f};
  v8f acc[8];
#pragma unroll
  for (int i = 0; i < 8; ++i) acc[i] = z;

  const int kcs = C >> 5;
  for (int s = 0; s < 9; ++s) {
    const int dy = s / 3, dx = s % 3;
    const int iy0 = oy0 + dy - pad, ix0 = ox0 + dx - pad;
    const int iy1 = oy1 + dy - pad, ix1 = ox1 + dx - pad;
    const bool ok0 = (iy0 >= 0) && (iy0 < Hin) && (ix0 >= 0) && (ix0 < Win);
    const bool ok1 = (iy1 >= 0) && (iy1 < Hin) && (ix1 >= 0) && (ix1 < Win);
    const __bf16* pin0 = inN + ((size_t)(iy0 * Win + ix0)) * C + kOff;
    const __bf16* pin1 = inN + ((size_t)(iy1 * Win + ix1)) * C + kOff;
    for (int kc = 0; kc < kcs; ++kc) {
      FragU a0, a1, b[4];
      v4u zz = {0u,0u,0u,0u};
      if (ok0) { const v4u* pa = (const v4u*)(pin0 + kc * 32); a0.u[0] = pa[0]; a0.u[1] = pa[1]; }
      else     { a0.u[0] = zz; a0.u[1] = zz; }
      if (ok1) { const v4u* pa = (const v4u*)(pin1 + kc * 32); a1.u[0] = pa[0]; a1.u[1] = pa[1]; }
      else     { a1.u[0] = zz; a1.u[1] = zz; }
      const __bf16* bbase = Wf + ((size_t)((s * kcs + kc) * ntiles)) * 512 + lane * 16;
#pragma unroll
      for (int t = 0; t < 4; ++t) {
        const v4u* pb = (const v4u*)(bbase + (size_t)(nt0 + t) * 512);
        b[t].u[0] = pb[0]; b[t].u[1] = pb[1];
      }
#pragma unroll
      for (int t = 0; t < 4; ++t) {
        acc[t]     = wmma_bf16(a0.v, b[t].v, acc[t]);
        acc[4 + t] = wmma_bf16(a1.v, b[t].v, acc[4 + t]);
      }
    }
  }

  const int col  = lane & 15;
  const int rOff = (lane >> 4) * 8;
  const size_t obase = (size_t)n * M * O;
#pragma unroll
  for (int h = 0; h < 2; ++h) {
#pragma unroll
    for (int t = 0; t < 4; ++t) {
      const int gn = nt0 * 16 + t * 16 + col;
#pragma unroll
      for (int r = 0; r < 8; ++r) {
        const int gm = mBase + h * 16 + rOff + r;
        if (gm < M) {
          size_t o = obase + (size_t)gm * O + gn;
          float v = acc[h * 4 + t][r];
          if (resid) v += resid[o];
          if (relu)  v = fmaxf(v, 0.f);
          if (outF) outF[o] = v;
          if (outB) outB[o] = f2bf(v);
        }
      }
    }
  }
}

// =====================================================================
// ROI-align (14x14 bilinear samples -> 2x2 mean -> 7x7 max) from NHWC feats
// one block per roi, threads over channels
// =====================================================================
__global__ void k_roi(const float* __restrict__ rois, const float* __restrict__ feats,
                      float* __restrict__ outF, __bf16* __restrict__ outB) {
  const int n = blockIdx.x;
  const float* rr = rois + n * 5;
  const int b = (int)rr[0];
  const float x1 = rr[1] * (float)WW, y1 = rr[2] * (float)HH;
  const float x2 = rr[3] * (float)WW, y2 = rr[4] * (float)HH;
  const float bw = fmaxf(x2 - x1, 1.f) * (1.f / 7.f);
  const float bh = fmaxf(y2 - y1, 1.f) * (1.f / 7.f);
  for (int c = threadIdx.x; c < RED; c += blockDim.x) {
    const float* fb = feats + (size_t)b * HW_ * RED + c;
    float mx = -3.4e38f;
    for (int by = 0; by < 7; ++by)
      for (int bx = 0; bx < 7; ++bx) {
        float s = 0.f;
#pragma unroll
        for (int sy = 0; sy < 2; ++sy)
#pragma unroll
          for (int sx = 0; sx < 2; ++sx) {
            float yv = y1 + ((float)by + ((float)sy + 0.5f) * 0.5f) * bh;
            float xv = x1 + ((float)bx + ((float)sx + 0.5f) * 0.5f) * bw;
            bool m = (yv >= -1.f) && (yv <= (float)HH) && (xv >= -1.f) && (xv <= (float)WW);
            float yc = fminf(fmaxf(yv, 0.f), (float)(HH - 1));
            float xc = fminf(fmaxf(xv, 0.f), (float)(WW - 1));
            int y0 = (int)floorf(yc);  int y1i = min(y0 + 1, HH - 1);
            int x0 = (int)floorf(xc);  int x1i = min(x0 + 1, WW - 1);
            float ly = yc - (float)y0, lx = xc - (float)x0;
            float v00 = fb[(size_t)(y0  * WW + x0 ) * RED];
            float v01 = fb[(size_t)(y0  * WW + x1i) * RED];
            float v10 = fb[(size_t)(y1i * WW + x0 ) * RED];
            float v11 = fb[(size_t)(y1i * WW + x1i) * RED];
            float val = v00 * (1 - ly) * (1 - lx) + v01 * (1 - ly) * lx
                      + v10 * ly * (1 - lx)       + v11 * ly * lx;
            s += m ? val : 0.f;
          }
        mx = fmaxf(mx, s * 0.25f);
      }
    outF[(size_t)n * RED + c] = mx;
    outB[(size_t)n * RED + c] = f2bf(mx);
  }
}

// inter1[n][hw][o] = relu(bg[b][hw][o] + actor[n][o]) -> bf16
__global__ void k_inter(const float* __restrict__ bgp, const float* __restrict__ act,
                        __bf16* __restrict__ out) {
  int idx = blockIdx.x * blockDim.x + threadIdx.x;
  if (idx >= NROI * HW_ * HID_) return;
  int o  = idx & (HID_ - 1);
  int t  = idx >> 9;
  int hw = t % HW_;
  int n  = t / HW_;
  int b  = n / RR;
  float v = bgp[((size_t)b * HW_ + hw) * HID_ + o] + act[(size_t)n * HID_ + o];
  out[idx] = f2bf(fmaxf(v, 0.f));
}

// 3x3/s2 maxpool (pad 1): (n,14,20,512) f32 -> (n,7,10,512) f32 + bf16
__global__ void k_pool(const float* __restrict__ cin, float* __restrict__ xf,
                       __bf16* __restrict__ xb) {
  int idx = blockIdx.x * blockDim.x + threadIdx.x;
  if (idx >= NROI * PHW_ * HID_) return;
  int o = idx & (HID_ - 1);
  int t = idx >> 9;
  int px = t % PW_;  t /= PW_;
  int py = t % PH_;
  int n  = t / PH_;
  float m = -3.4e38f;
#pragma unroll
  for (int ky = 0; ky < 3; ++ky) {
    int y = 2 * py - 1 + ky;
    if (y < 0 || y >= CH_) continue;
#pragma unroll
    for (int kx = 0; kx < 3; ++kx) {
      int x = 2 * px - 1 + kx;
      if (x < 0 || x >= CW_) continue;
      m = fmaxf(m, cin[(((size_t)n * CH_ + y) * CW_ + x) * HID_ + o]);
    }
  }
  xf[idx] = m;
  xb[idx] = f2bf(m);
}

// HR2O attention: per (batch, pixel): att = softmax_j(q_i . k_j / sqrt(512));
// virt_i = sum_j att_ij * v_j    (all f32, NHWC [n][70][512])
__global__ void k_attn(const float* __restrict__ q, const float* __restrict__ k,
                       const float* __restrict__ v, float* __restrict__ virt) {
  const int bp  = blockIdx.x;          // 0..559
  const int b   = bp / PHW_;
  const int pos = bp % PHW_;
  __shared__ float att[RR][RR];
  for (int pr = threadIdx.x; pr < RR * RR; pr += blockDim.x) {
    int i = pr / RR, j = pr % RR;
    const float* qi = q + ((size_t)(b * RR + i) * PHW_ + pos) * HID_;
    const float* kj = k + ((size_t)(b * RR + j) * PHW_ + pos) * HID_;
    float d = 0.f;
    for (int c = 0; c < HID_; ++c) d += qi[c] * kj[c];
    att[i][j] = d * 0.04419417382415922f;     // 1/sqrt(512)
  }
  __syncthreads();
  if (threadIdx.x < RR) {
    int i = threadIdx.x;
    float m = -3.4e38f;
    for (int j = 0; j < RR; ++j) m = fmaxf(m, att[i][j]);
    float e[RR]; float s = 0.f;
    for (int j = 0; j < RR; ++j) { e[j] = __expf(att[i][j] - m); s += e[j]; }
    float inv = 1.f / s;
    for (int j = 0; j < RR; ++j) att[i][j] = e[j] * inv;
  }
  __syncthreads();
  for (int idx = threadIdx.x; idx < RR * HID_; idx += blockDim.x) {
    int i = idx >> 9, c = idx & (HID_ - 1);
    float s = 0.f;
#pragma unroll
    for (int j = 0; j < RR; ++j)
      s += att[i][j] * v[((size_t)(b * RR + j) * PHW_ + pos) * HID_ + c];
    virt[((size_t)(b * RR + i) * PHW_ + pos) * HID_ + c] = s;
  }
}

// layer-norm over (C,H,W) per sample + per-channel affine + relu -> bf16
__global__ void k_lnrelu(const float* __restrict__ virt, const float* __restrict__ g,
                         const float* __restrict__ bia, __bf16* __restrict__ out) {
  const int n = blockIdx.x;
  const int NE = PHW_ * HID_;   // 35840
  const float* p = virt + (size_t)n * NE;
  float s = 0.f, s2 = 0.f;
  for (int e = threadIdx.x; e < NE; e += 256) { float x = p[e]; s += x; s2 += x * x; }
  __shared__ float sh1[256], sh2[256];
  sh1[threadIdx.x] = s; sh2[threadIdx.x] = s2;
  __syncthreads();
  for (int o = 128; o > 0; o >>= 1) {
    if (threadIdx.x < o) { sh1[threadIdx.x] += sh1[threadIdx.x + o];
                           sh2[threadIdx.x] += sh2[threadIdx.x + o]; }
    __syncthreads();
  }
  const float mu   = sh1[0] / (float)NE;
  const float var  = sh2[0] / (float)NE - mu * mu;
  const float rstd = rsqrtf(var + 1e-5f);
  for (int e = threadIdx.x; e < NE; e += 256) {
    int c = e & (HID_ - 1);
    float x = (p[e] - mu) * rstd * g[c] + bia[c];
    out[(size_t)n * NE + e] = f2bf(fmaxf(x, 0.f));
  }
}

// hof[n][c] = mean over hw -> bf16 into concat[:,512:]
__global__ void k_hof(const float* __restrict__ xb, __bf16* __restrict__ concat) {
  const int n = blockIdx.x;
  for (int c = threadIdx.x; c < HID_; c += blockDim.x) {
    float s = 0.f;
    for (int hw = 0; hw < PHW_; ++hw) s += xb[((size_t)n * PHW_ + hw) * HID_ + c];
    concat[(size_t)n * (2 * HID_) + HID_ + c] = f2bf(s * (1.f / (float)PHW_));
  }
}

// =====================================================================
// host launcher
// =====================================================================
extern "C" void kernel_launch(void* const* d_in, const int* in_sizes, int n_in,
                              void* d_out, int out_size, void* d_ws, size_t ws_size,
                              hipStream_t stream) {
  const float* x        = (const float*)d_in[0];
  const float* rois     = (const float*)d_in[1];
  const float* w_reduce = (const float*)d_in[2];
  const float* w_conv1  = (const float*)d_in[3];
  const float* w_conv2  = (const float*)d_in[4];
  const float* w_q      = (const float*)d_in[5];
  const float* w_k      = (const float*)d_in[6];
  const float* w_v      = (const float*)d_in[7];
  const float* w_o      = (const float*)d_in[8];
  const float* gn_g     = (const float*)d_in[9];
  const float* gn_b     = (const float*)d_in[10];
  const float* w_fc1    = (const float*)d_in[11];
  const float* w_fc2    = (const float*)d_in[12];
  float* out_cls = (float*)d_out;                 // 80*80
  float* out_roi = (float*)d_out + NCLS_ * NROI;  // 80*1024

  // bump allocator on workspace (256B aligned)
  char* p = (char*)d_ws;
  auto bumpB  = [&](size_t bytes) -> void* {
    void* r = (void*)p; p += (bytes + 255) & ~((size_t)255); return r; };
  auto bumpBF = [&](size_t n) -> __bf16* { return (__bf16*)bumpB(n * 2); };
  auto bumpF  = [&](size_t n) -> float*  { return (float*)bumpB(n * 4); };

  const size_t TAPFRAG = (size_t)(HID_/32) * (HID_/16) * 512;    // 262144 per tap
  __bf16* fr_reduce = bumpBF((size_t)(WIDTH_/32) * (RED/16) * 512);  // 2048x1024
  __bf16* fr_bg     = bumpBF((size_t)(RED/32) * (HID_/16) * 512);
  __bf16* fr_act    = bumpBF((size_t)(RED/32) * (HID_/16) * 512);
  __bf16* fr_c2     = bumpBF(9 * TAPFRAG);
  __bf16* fr_q[2], *fr_k[2], *fr_v[2], *fr_o[2];
  for (int d = 0; d < 2; ++d) {
    fr_q[d] = bumpBF(9 * TAPFRAG); fr_k[d] = bumpBF(9 * TAPFRAG);
    fr_v[d] = bumpBF(9 * TAPFRAG); fr_o[d] = bumpBF(9 * TAPFRAG);
  }
  __bf16* fr_fc1 = bumpBF((size_t)(RED/32) * (HID_/16) * 512);
  __bf16* fr_fc2 = bumpBF((size_t)(RED/32) * 8 * 512);           // N=80 padded to 128

  __bf16* xm      = bumpBF((size_t)BB * HW_ * WIDTH_);
  float*  featsF  = bumpF((size_t)BB * HW_ * RED);
  __bf16* featsB  = bumpBF((size_t)BB * HW_ * RED);
  __bf16* roiB    = bumpBF((size_t)NROI * RED);
  float*  actorF  = bumpF((size_t)NROI * HID_);
  float*  bgp     = bumpF((size_t)BB * HW_ * HID_);
  __bf16* inter1  = bumpBF((size_t)NROI * HW_ * HID_);
  float*  convout = bumpF((size_t)NROI * CHW_ * HID_);
  float*  xbF[2]; __bf16* xbB[2];
  for (int d = 0; d < 2; ++d) {
    xbF[d] = bumpF((size_t)NROI * PHW_ * HID_);
    xbB[d] = bumpBF((size_t)NROI * PHW_ * HID_);
  }
  float*  qbuf   = bumpF((size_t)NROI * PHW_ * HID_);
  float*  kbuf   = bumpF((size_t)NROI * PHW_ * HID_);
  float*  vbuf   = bumpF((size_t)NROI * PHW_ * HID_);
  float*  virt   = bumpF((size_t)NROI * PHW_ * HID_);
  __bf16* lnrelu = bumpBF((size_t)NROI * PHW_ * HID_);
  __bf16* concat = bumpBF((size_t)NROI * 2 * HID_);

  auto prep = [&](const float* src, __bf16* dst, int K, int N, int ntiles, int sk, int sn) {
    int total = (K / 32) * ntiles * 512;
    k_prep_frag<<<dim3((total + 255) / 256), dim3(256), 0, stream>>>(src, dst, K, N, ntiles, sk, sn);
  };
  auto gemm = [&](const __bf16* A, const __bf16* Bf, float* Cf, __bf16* Cb,
                  int M, int N, int K, int lda, int ldc, int ntiles, int relu) {
    k_gemm<<<dim3((M + 127) / 128, (ntiles + 3) / 4), dim3(128), 0, stream>>>(
        A, Bf, Cf, Cb, M, N, K, lda, ldc, ntiles, relu);
  };
  auto conv = [&](const __bf16* in, const __bf16* Wf, float* oF, __bf16* oB, const float* res,
                  int Hin, int Win, int Hout, int Wout, int pad, int relu) {
    int M = Hout * Wout;
    k_conv3x3<<<dim3((M + 31) / 32, HID_ / 64, NROI), dim3(32), 0, stream>>>(
        in, Wf, oF, oB, res, Hin, Win, HID_, Hout, Wout, HID_, HID_ / 16, pad, relu);
  };

  // ---- pre-swizzle all weight matrices into WMMA B-fragment layout ----
  prep(w_reduce, fr_reduce, WIDTH_, RED, RED / 16, 1, WIDTH_);           // W^T
  prep(w_conv1,        fr_bg,  RED, HID_, HID_ / 16, 1, 2 * RED);
  prep(w_conv1 + RED,  fr_act, RED, HID_, HID_ / 16, 1, 2 * RED);
  for (int s = 0; s < 9; ++s)
    prep(w_conv2 + s, fr_c2 + (size_t)s * TAPFRAG, HID_, HID_, HID_ / 16, 9, HID_ * 9);
  const size_t WD = (size_t)HID_ * HID_ * 9;
  for (int d = 0; d < 2; ++d)
    for (int s = 0; s < 9; ++s) {
      prep(w_q + d * WD + s, fr_q[d] + (size_t)s * TAPFRAG, HID_, HID_, HID_ / 16, 9, HID_ * 9);
      prep(w_k + d * WD + s, fr_k[d] + (size_t)s * TAPFRAG, HID_, HID_, HID_ / 16, 9, HID_ * 9);
      prep(w_v + d * WD + s, fr_v[d] + (size_t)s * TAPFRAG, HID_, HID_, HID_ / 16, 9, HID_ * 9);
      prep(w_o + d * WD + s, fr_o[d] + (size_t)s * TAPFRAG, HID_, HID_, HID_ / 16, 9, HID_ * 9);
    }
  prep(w_fc1, fr_fc1, RED, HID_, HID_ / 16, 1, RED);
  prep(w_fc2, fr_fc2, RED, NCLS_, 8, 1, RED);                            // pad N 80->128

  // ---- pipeline ----
  k_meanT<<<dim3(HW_ / 32, WIDTH_ / 64, BB), dim3(256), 0, stream>>>(x, xm);

  // feats = xm @ w_reduce^T  : [2816 x 2048] x [2048 x 1024]
  gemm(xm, fr_reduce, featsF, featsB, BB * HW_, RED, WIDTH_, WIDTH_, RED, RED / 16, 0);

  // roi-align + max -> out_roi (f32, second output) + bf16
  k_roi<<<dim3(NROI), dim3(256), 0, stream>>>(rois, featsF, out_roi, roiB);

  // actor bias: [80 x 1024] x [1024 x 512]
  gemm(roiB, fr_act, actorF, nullptr, NROI, HID_, RED, RED, HID_, HID_ / 16, 0);
  // bg (shared across rois of a batch): [2816 x 1024] x [1024 x 512]
  gemm(featsB, fr_bg, bgp, nullptr, BB * HW_, HID_, RED, RED, HID_, HID_ / 16, 0);
  {
    int total = NROI * HW_ * HID_;
    k_inter<<<dim3((total + 255) / 256), dim3(256), 0, stream>>>(bgp, actorF, inter1);
  }
  // conv2 3x3 VALID + relu : (80,16,22,512) -> (80,14,20,512)
  conv(inter1, fr_c2, convout, nullptr, nullptr, HH, WW, CH_, CW_, 0, 1);
  {
    int total = NROI * PHW_ * HID_;
    k_pool<<<dim3((total + 255) / 256), dim3(256), 0, stream>>>(convout, xbF[0], xbB[0]);
  }

  // ---- 2x HR2O blocks ----
  int cur = 0;
  for (int d = 0; d < 2; ++d) {
    int nxt = 1 - cur;
    conv(xbB[cur], fr_q[d], qbuf, nullptr, nullptr, PH_, PW_, PH_, PW_, 1, 0);
    conv(xbB[cur], fr_k[d], kbuf, nullptr, nullptr, PH_, PW_, PH_, PW_, 1, 0);
    conv(xbB[cur], fr_v[d], vbuf, nullptr, nullptr, PH_, PW_, PH_, PW_, 1, 0);
    k_attn<<<dim3(BB * PHW_), dim3(128), 0, stream>>>(qbuf, kbuf, vbuf, virt);
    k_lnrelu<<<dim3(NROI), dim3(256), 0, stream>>>(virt, gn_g + d * HID_, gn_b + d * HID_, lnrelu);
    conv(lnrelu, fr_o[d], xbF[nxt], xbB[nxt], xbF[cur], PH_, PW_, PH_, PW_, 1, 0);
    cur = nxt;
  }

  // ---- heads ----
  k_hof<<<dim3(NROI), dim3(256), 0, stream>>>(xbF[cur], concat);
  // out = relu(roi @ fc1^T) into concat[:, :512] (row stride 1024)
  gemm(roiB, fr_fc1, nullptr, concat, NROI, HID_, RED, RED, 2 * HID_, HID_ / 16, 1);
  // cls = concat @ fc2^T -> d_out[0:6400]
  gemm(concat, fr_fc2, out_cls, nullptr, NROI, NCLS_, RED, 2 * HID_, NCLS_, 8, 0);

  (void)in_sizes; (void)n_in; (void)out_size; (void)ws_size;
}
